// TSegFormer_22986664968414
// MI455X (gfx1250) — matrix operand order repaired and checked by reading, not compiled, not run on hardware
//
#include <hip/hip_runtime.h>

#define BN_INV 0.99999500003749978f  // 1/sqrt(1+1e-5)

typedef _Float16 v16h __attribute__((ext_vector_type(16)));
typedef _Float16 v8h  __attribute__((ext_vector_type(8)));
typedef _Float16 v4h  __attribute__((ext_vector_type(4)));
typedef float    v8f  __attribute__((ext_vector_type(8)));

// ---------------------------------------------------------------------------
// Strided batched GEMM with fused epilogue, LDS-staged tiles.
// REQUIRES: K % 32 == 0, B row-contiguous (stride bK, bK % 4 == 0),
//           N % 256 == 0 (true for all call sites: N = 2048).
//   C[b,m,n] = act( (alpha*sum_k A[b,m,k]*B[b,k,n] + bias[m]) * (gamma[m]*BN_INV) + beta[m] ) + resid[b,m,n]
// Block = 128 thr (4 waves) computes a 16(M) x 256(N) tile; per 32-k step the
// block stages A(16x32) and B(32x256, k-pair interleaved) to LDS as f16, then
// each wave runs 4x v_wmma_f32_16x16x32_f16 on its 64-column strip.
// B LDS layout: sB2[j][n] = {B[2j][n], B[2j+1][n]} so a lane's fragment pair
// is one aligned 32-bit LDS word (ds_load_b32, conflict-free across lanes).
// ---------------------------------------------------------------------------
__global__ __launch_bounds__(128) void gemm_bn_act_kernel(
    const float* __restrict__ A, long aB, long aM, long aK,
    const float* __restrict__ Bm, long bB, long bK,
    float* __restrict__ C, long cB,
    const float* __restrict__ bias,
    const float* __restrict__ gamma,
    const float* __restrict__ beta,
    const float* __restrict__ resid, long rB,
    int M, int N, int K, float alpha, int act)
{
  __shared__ __align__(16) _Float16 sA[16][40];       // [m][k], padded row
  __shared__ __align__(16) _Float16 sB2[16][260][2];  // [k/2][n][pair], padded

  int tid   = threadIdx.x;
  int wave  = tid >> 5;
  int lane  = tid & 31;
  int lhalf = lane >> 4;
  int l15   = lane & 15;
  int n0 = blockIdx.x * 256;
  int m0 = blockIdx.y * 16;
  int b  = blockIdx.z;

  const float* Ab = A + (long)b * aB;
  const float* Bb = Bm + (long)b * bB;

  v8f acc[4];
#pragma unroll
  for (int t = 0; t < 4; ++t) acc[t] = (v8f){0.f,0.f,0.f,0.f,0.f,0.f,0.f,0.f};

  const int kaoff = lhalf ? 16 : 0;  // A-frag lane K offset (contiguous k = 16h + i)
  const int jboff = lhalf ? 8 : 0;   // B-frag lane K-pair offset (k base 0 / 16)

  // staging decomposition
  int arow = tid >> 5;        // A: thread covers (arow + 4j, acol)
  int acol = tid & 31;
  int bsub = tid >> 6;        // B: pair-rows jj*2 + bsub
  int bnq  = tid & 63;        // B: columns bnq*4 .. +3

  for (int k0 = 0; k0 < K; k0 += 32) {
    __syncthreads();
    // ---- stage A tile (16x32) ----
#pragma unroll
    for (int j = 0; j < 4; ++j) {
      int r = arow + j * 4;
      int m = m0 + r; if (m > M - 1) m = M - 1;   // clamp; masked at store
      sA[r][acol] = (_Float16)Ab[(long)m * aM + (long)(k0 + acol) * aK];
    }
    // ---- stage B tile (32x256) with float4 coalesced loads, pair-packed ----
#pragma unroll
    for (int jj = 0; jj < 8; ++jj) {
      int jp = jj * 2 + bsub;  // pair-row 0..15 -> k = 2jp, 2jp+1
      const float4 v0 = *reinterpret_cast<const float4*>(
          Bb + (long)(k0 + 2 * jp) * bK + (n0 + bnq * 4));
      const float4 v1 = *reinterpret_cast<const float4*>(
          Bb + (long)(k0 + 2 * jp + 1) * bK + (n0 + bnq * 4));
      v8h pk = { (_Float16)v0.x, (_Float16)v1.x, (_Float16)v0.y, (_Float16)v1.y,
                 (_Float16)v0.z, (_Float16)v1.z, (_Float16)v0.w, (_Float16)v1.w };
      *reinterpret_cast<v8h*>(&sB2[jp][bnq * 4][0]) = pk;  // 16B aligned (260*4B rows)
    }
    __syncthreads();

    // ---- fragments from LDS + 4 WMMAs per wave ----
    v16h af;
#pragma unroll
    for (int i = 0; i < 16; ++i) {
      int ka = i + kaoff;
      af[i] = sA[l15][ka];
    }
#pragma unroll
    for (int t = 0; t < 4; ++t) {
      int nloc = wave * 64 + t * 16 + l15;
      v16h bf;
#pragma unroll
      for (int j = 0; j < 8; ++j) {
        int jp = jboff + j;              // k = 2*jp, 2*jp+1
        bf[2 * j]     = sB2[jp][nloc][0];
        bf[2 * j + 1] = sB2[jp][nloc][1];
      }
      acc[t] = __builtin_amdgcn_wmma_f32_16x16x32_f16(false, af, false, bf,
                                                      (short)0, acc[t], false, false);
    }
  }

#pragma unroll
  for (int t = 0; t < 4; ++t) {
#pragma unroll
    for (int r = 0; r < 8; ++r) {
      int m = m0 + r + lhalf * 8;   // C/D layout: VGPR r -> M=r (lanes 0-15) / M=r+8 (lanes 16-31)
      int n = n0 + wave * 64 + t * 16 + l15;
      if (m < M && n < N) {
        float v = acc[t][r] * alpha;
        if (bias)  v += bias[m];
        if (gamma) v = v * (gamma[m] * BN_INV) + (beta ? beta[m] : 0.f);
        if (act == 1)      v = fmaxf(v, 0.f);
        else if (act == 2) v = (v > 0.f) ? v : 0.2f * v;
        if (resid) v += resid[(long)b * rB + (long)m * N + n];
        C[(long)b * cB + (long)m * N + n] = v;
      }
    }
  }
}

// ---------------------------------------------------------------------------
// First conv (w_c1: 64x7, K=7 not a multiple of 32) as a plain VALU kernel:
//   out[b][o][n] = relu(bn(sum_c w[o][c]*x[b][c][n]))
// ---------------------------------------------------------------------------
__global__ void conv7_bn_relu_kernel(const float* __restrict__ x, const float* __restrict__ w,
                                     const float* __restrict__ g, const float* __restrict__ bb,
                                     float* __restrict__ out, int N)
{
  int t = blockIdx.x * blockDim.x + threadIdx.x;  // over 64*N
  int b = blockIdx.y;
  if (t >= 64 * N) return;
  int o = t / N, n = t - o * N;
  const float* xb = x + (long)b * 7 * N;
  float s = 0.f;
#pragma unroll
  for (int c = 0; c < 7; ++c) s += w[o * 7 + c] * xb[(long)c * N + n];
  s = s * (g[o] * BN_INV) + bb[o];
  out[(long)b * 64 * N + t] = fmaxf(s, 0.f);
}

// ---------------------------------------------------------------------------
// KNN: 32 nearest neighbors per point (set-equality is enough: downstream max
// over neighbors is order-invariant). One thread per (b,n).
// ---------------------------------------------------------------------------
__global__ void knn_kernel(const float* __restrict__ x, int* __restrict__ idx, int N)
{
  int n = blockIdx.x * blockDim.x + threadIdx.x;
  int b = blockIdx.y;
  if (n >= N) return;
  const float* xb = x + (long)b * 7 * N;
  float px = xb[n], py = xb[N + n], pz = xb[2 * N + n];
  float bd[32];
  int   bi[32];
#pragma unroll
  for (int i = 0; i < 32; ++i) { bd[i] = 3.4e38f; bi[i] = 0; }
  for (int m = 0; m < N; ++m) {
    float dx = xb[m] - px, dy = xb[N + m] - py, dz = xb[2 * N + m] - pz;
    float d = dx * dx + dy * dy + dz * dz;
    if (d < bd[31]) {
      int j = 31;
      while (j > 0 && bd[j - 1] > d) { bd[j] = bd[j - 1]; bi[j] = bi[j - 1]; --j; }
      bd[j] = d; bi[j] = m;
    }
  }
  int* o = idx + ((long)b * N + n) * 32;
  for (int i = 0; i < 32; ++i) o[i] = bi[i];
}

// ---------------------------------------------------------------------------
// Fused sample_and_group + local_op. One workgroup (256 thr = 8 waves) per
// point. feat(128x32), h1, h2 live in LDS; both GEMMs use WMMA.
//   feat[d][s] = d<64 ? p[idx[s]][d]-p[n][d] : p[n][d-64]
//   h1 = relu(bn(w1(OC1x128) @ feat)) ; h2 = relu(bn(w2(OCxOC1) @ h1))
//   out[b][o][n] = max_s h2[o][s]
// ---------------------------------------------------------------------------
__global__ __launch_bounds__(256) void localop_kernel(
    const float* __restrict__ pts, const int* __restrict__ idx,
    const float* __restrict__ w1, const float* __restrict__ g1, const float* __restrict__ b1,
    const float* __restrict__ w2, const float* __restrict__ g2, const float* __restrict__ b2,
    float* __restrict__ out, int OC1, int OC, int N)
{
  __shared__ float sfeat[128][32];
  __shared__ float sh1[128][32];
  __shared__ float sh2[128][32];
  __shared__ int   sidx[32];

  int n = blockIdx.x, b = blockIdx.y;
  const float* pb = pts + (long)b * 64 * N;

  if (threadIdx.x < 32)
    sidx[threadIdx.x] = idx[((long)b * N + n) * 32 + threadIdx.x];
  __syncthreads();

  for (int e = threadIdx.x; e < 128 * 32; e += blockDim.x) {
    int d = e >> 5, s = e & 31;
    float v;
    if (d < 64) v = pb[(long)d * N + sidx[s]] - pb[(long)d * N + n];
    else        v = pb[(long)(d - 64) * N + n];
    sfeat[d][s] = v;
  }
  __syncthreads();

  int wave  = threadIdx.x >> 5;
  int lane  = threadIdx.x & 31;
  int lhalf = lane >> 4;
  int l15   = lane & 15;
  const int kaoff = lhalf ? 16 : 0;
  const int kboff = lhalf ? 16 : 0;

  // GEMM1: h1[o][s] = sum_d w1[o][d]*feat[d][s]  (K=128)
  int ntasks1 = (OC1 >> 4) * 2;
  for (int task = wave; task < ntasks1; task += 8) {
    int m0 = (task >> 1) * 16, s0 = (task & 1) * 16;
    const float* wrow = w1 + (long)(m0 + l15) * 128;
    v8f acc = {0.f,0.f,0.f,0.f,0.f,0.f,0.f,0.f};
    for (int k0 = 0; k0 < 128; k0 += 32) {
      v16h af, bf;
#pragma unroll
      for (int i = 0; i < 16; ++i) {
        int ka = k0 + i + kaoff;
        af[i] = (_Float16)wrow[ka];
        int kb = k0 + i + kboff;
        bf[i] = (_Float16)sfeat[kb][s0 + l15];
      }
      acc = __builtin_amdgcn_wmma_f32_16x16x32_f16(false, af, false, bf,
                                                   (short)0, acc, false, false);
    }
#pragma unroll
    for (int r = 0; r < 8; ++r) {
      int m = m0 + r + lhalf * 8, s = s0 + l15;
      float t = acc[r] * (g1[m] * BN_INV) + b1[m];
      sh1[m][s] = fmaxf(t, 0.f);
    }
  }
  __syncthreads();

  // GEMM2: h2[o][s] = sum_c w2[o][c]*h1[c][s]  (K=OC1, multiple of 32)
  int ntasks2 = (OC >> 4) * 2;
  for (int task = wave; task < ntasks2; task += 8) {
    int m0 = (task >> 1) * 16, s0 = (task & 1) * 16;
    const float* wrow = w2 + (long)(m0 + l15) * OC1;
    v8f acc = {0.f,0.f,0.f,0.f,0.f,0.f,0.f,0.f};
    for (int k0 = 0; k0 < OC1; k0 += 32) {
      v16h af, bf;
#pragma unroll
      for (int i = 0; i < 16; ++i) {
        int ka = k0 + i + kaoff;
        af[i] = (_Float16)wrow[ka];
        int kb = k0 + i + kboff;
        bf[i] = (_Float16)sh1[kb][s0 + l15];
      }
      acc = __builtin_amdgcn_wmma_f32_16x16x32_f16(false, af, false, bf,
                                                   (short)0, acc, false, false);
    }
#pragma unroll
    for (int r = 0; r < 8; ++r) {
      int m = m0 + r + lhalf * 8, s = s0 + l15;
      float t = acc[r] * (g2[m] * BN_INV) + b2[m];
      sh2[m][s] = fmaxf(t, 0.f);
    }
  }
  __syncthreads();

  for (int o = threadIdx.x; o < OC; o += blockDim.x) {
    float mv = sh2[o][0];
    for (int s = 1; s < 32; ++s) mv = fmaxf(mv, sh2[o][s]);
    out[(long)b * OC * N + (long)o * N + n] = mv;
  }
}

// ---------------------------------------------------------------------------
// In-place softmax over axis n (rows) of energy[b][n][m], for each column m.
// Threads over m -> coalesced row reads.
// ---------------------------------------------------------------------------
__global__ void softmax_col_kernel(float* __restrict__ e, int N)
{
  int m = blockIdx.x * blockDim.x + threadIdx.x;
  int b = blockIdx.y;
  if (m >= N) return;
  float* p = e + (long)b * N * N;
  float mx = -3.4e38f;
  for (int n = 0; n < N; ++n) mx = fmaxf(mx, p[(long)n * N + m]);
  float s = 0.f;
  for (int n = 0; n < N; ++n) s += __expf(p[(long)n * N + m] - mx);
  float inv = 1.f / s;
  for (int n = 0; n < N; ++n) {
    long a = (long)n * N + m;
    p[a] = __expf(p[a] - mx) * inv;
  }
}

// max & mean over N for each (b,c)
__global__ void pool_kernel(const float* __restrict__ xx, float* __restrict__ xmax,
                            float* __restrict__ xavg, int C, int N)
{
  int c = blockIdx.x, b = blockIdx.y;
  const float* p = xx + ((long)b * C + c) * N;
  float mx = -3.4e38f, sm = 0.f;
  for (int n = threadIdx.x; n < N; n += blockDim.x) {
    float v = p[n];
    mx = fmaxf(mx, v);
    sm += v;
  }
  __shared__ float smx[256], ssm[256];
  smx[threadIdx.x] = mx; ssm[threadIdx.x] = sm;
  __syncthreads();
  for (int s = 128; s > 0; s >>= 1) {
    if (threadIdx.x < s) {
      smx[threadIdx.x] = fmaxf(smx[threadIdx.x], smx[threadIdx.x + s]);
      ssm[threadIdx.x] += ssm[threadIdx.x + s];
    }
    __syncthreads();
  }
  if (threadIdx.x == 0) {
    xmax[(long)b * C + c] = smx[0];
    xavg[(long)b * C + c] = ssm[0] / (float)N;
  }
}

// clf[b][o] = lrelu(bn(w_lab(64x2) @ cls[b])) ; 256 threads total (B=4, 64 ch)
__global__ void clf_kernel(const float* __restrict__ cls, const float* __restrict__ wl,
                           const float* __restrict__ g, const float* __restrict__ bb,
                           float* __restrict__ clf)
{
  int t = threadIdx.x;
  int b = t >> 6, o = t & 63;
  float v = wl[o * 2] * cls[b * 2] + wl[o * 2 + 1] * cls[b * 2 + 1];
  v = v * (g[o] * BN_INV) + bb[o];
  v = (v > 0.f) ? v : 0.2f * v;
  clf[b * 64 + o] = v;
}

// all_fea[b][ch][n] = concat(xx(1024), bcast xmax(1024), bcast xavg(1024), bcast clf(64))
__global__ void allfea_kernel(const float* __restrict__ xx, const float* __restrict__ xmax,
                              const float* __restrict__ xavg, const float* __restrict__ clf,
                              float* __restrict__ out, int N)
{
  long e = (long)blockIdx.x * blockDim.x + threadIdx.x;
  long total = (long)4 * 3136 * N;
  if (e >= total) return;
  int n = (int)(e % N);
  long r = e / N;
  int ch = (int)(r % 3136);
  int b = (int)(r / 3136);
  float v;
  if (ch < 1024)      v = xx[((long)b * 1024 + ch) * N + n];
  else if (ch < 2048) v = xmax[b * 1024 + (ch - 1024)];
  else if (ch < 3072) v = xavg[b * 1024 + (ch - 2048)];
  else                v = clf[b * 64 + (ch - 3072)];
  out[e] = v;
}

// ---------------------------------------------------------------------------
extern "C" void kernel_launch(void* const* d_in, const int* in_sizes, int n_in,
                              void* d_out, int out_size, void* d_ws, size_t ws_size,
                              hipStream_t stream)
{
  (void)in_sizes; (void)n_in; (void)out_size; (void)ws_size;
  const int Bn = 4, N = 2048;
  const float* x   = (const float*)d_in[0];
  const float* cls = (const float*)d_in[1];
  auto P = [&](int i) { return (const float*)d_in[i]; };

  // ----- workspace layout (floats) -----
  float* W = (float*)d_ws;
  size_t off = 0;
  auto alloc = [&](size_t nf) { size_t r = off; off += nf; return r; };
  float* buf_h    = W + alloc((size_t)Bn * 64 * N);
  float* buf_pts  = W + alloc((size_t)Bn * 64 * N);
  float* buf_f0   = W + alloc((size_t)Bn * 64 * N);
  float* buf_f1   = W + alloc((size_t)Bn * 128 * N);
  float* buf_qk   = W + alloc((size_t)Bn * 32 * N);
  float* buf_v    = W + alloc((size_t)Bn * 128 * N);
  float* buf_xr   = W + alloc((size_t)Bn * 128 * N);
  float* buf_sa   = W + alloc((size_t)Bn * 512 * N);
  float* buf_xx   = W + alloc((size_t)Bn * 1024 * N);
  float* buf_xmax = W + alloc((size_t)Bn * 1024);
  float* buf_xavg = W + alloc((size_t)Bn * 1024);
  float* buf_clf  = W + alloc((size_t)Bn * 64);
  int*   buf_idx  = (int*)(W + alloc((size_t)Bn * N * 32));
  float* buf_big  = W + alloc((size_t)Bn * 3136 * N);   // energy (B*N*N fits) then all_fea
  float* buf_t1   = W + alloc((size_t)Bn * 512 * N);
  float* buf_t2   = W + alloc((size_t)Bn * 256 * N);

  auto gemm = [&](const float* A, long aB, long aM, long aK,
                  const float* Bmat, long bB, long bK,
                  float* C, long cB,
                  const float* bias, const float* gamma, const float* beta,
                  const float* resid, long rB,
                  int M, int Nn, int K, float alpha, int act) {
    dim3 grid(Nn / 256, (M + 15) / 16, Bn);
    gemm_bn_act_kernel<<<grid, 128, 0, stream>>>(
        A, aB, aM, aK, Bmat, bB, bK, C, cB,
        bias, gamma, beta, resid, rB, M, Nn, K, alpha, act);
  };

  // 1. KNN on xyz (first 3 channels of x)
  knn_kernel<<<dim3(N / 128, Bn), 128, 0, stream>>>(x, buf_idx, N);

  // 2. h = relu(bn(w_c1 @ x))  (K=7 -> dedicated VALU kernel)
  conv7_bn_relu_kernel<<<dim3((64 * N + 255) / 256, Bn), 256, 0, stream>>>(
      x, P(2), P(3), P(4), buf_h, N);
  // 3. pts = relu(bn(w_c2 @ h))      w_c2: 64x64
  gemm(P(5), 0, 64, 1, buf_h, (long)64 * N, N, buf_pts, (long)64 * N,
       nullptr, P(6), P(7), nullptr, 0, 64, N, 64, 1.f, 1);

  // 4. f0 = local_op level 0 (OC1=64, OC=64)
  localop_kernel<<<dim3(N, Bn), 256, 0, stream>>>(
      buf_pts, buf_idx, P(8), P(9), P(10), P(11), P(12), P(13), buf_f0, 64, 64, N);
  // 5. f1 = local_op level 1 (OC1=128, OC=128)
  localop_kernel<<<dim3(N, Bn), 256, 0, stream>>>(
      buf_f0, buf_idx, P(14), P(15), P(16), P(17), P(18), P(19), buf_f1, 128, 128, N);

  // 6. four SA (offset attention) layers
  const float inv_sqrt_c4 = 0.17677669529663687f;  // 1/sqrt(32)
  for (int i = 0; i < 4; ++i) {
    int base = 20 + i * 7;
    const float* wqk = P(base + 0);
    const float* wv  = P(base + 1);
    const float* bv  = P(base + 2);
    const float* wt  = P(base + 3);
    const float* bt  = P(base + 4);
    const float* g   = P(base + 5);
    const float* bb  = P(base + 6);
    const float* xin; long xinB;
    if (i == 0) { xin = buf_f1; xinB = (long)128 * N; }
    else        { xin = buf_sa + (size_t)(i - 1) * 128 * N; xinB = (long)512 * N; }

    // qk = wqk @ xin  (32 x N)
    gemm(wqk, 0, 128, 1, xin, xinB, N, buf_qk, (long)32 * N,
         nullptr, nullptr, nullptr, nullptr, 0, 32, N, 128, 1.f, 0);
    // v = wv @ xin + bv
    gemm(wv, 0, 128, 1, xin, xinB, N, buf_v, (long)128 * N,
         bv, nullptr, nullptr, nullptr, 0, 128, N, 128, 1.f, 0);
    // energy[n][m] = (1/sqrt(32)) * sum_o qk[o][n] * qk[o][m]   (A: aM=1, aK=N)
    gemm(buf_qk, (long)32 * N, 1, N,  buf_qk, (long)32 * N, N,
         buf_big, (long)N * N, nullptr, nullptr, nullptr, nullptr, 0,
         N, N, 32, inv_sqrt_c4, 0);
    // softmax over n (axis -2), in place
    softmax_col_kernel<<<dim3(N / 256, Bn), 256, 0, stream>>>(buf_big, N);
    // x_r[c][m] = sum_n v[c][n] * attn[n][m]
    gemm(buf_v, (long)128 * N, N, 1, buf_big, (long)N * N, N,
         buf_xr, (long)128 * N, nullptr, nullptr, nullptr, nullptr, 0,
         128, N, N, 1.f, 0);
    // out = xin + relu(bn(wt @ x_r + bt)) -> SA output slice i
    float* outp = buf_sa + (size_t)i * 128 * N;
    gemm(wt, 0, 128, 1, buf_xr, (long)128 * N, N, outp, (long)512 * N,
         bt, g, bb, xin, xinB, 128, N, 128, 1.f, 1);
  }

  // 7. xx = lrelu(bn(w_fuse @ concat(sa1..sa4)))   w_fuse: 1024x512
  gemm(P(48), 0, 512, 1, buf_sa, (long)512 * N, N, buf_xx, (long)1024 * N,
       nullptr, P(49), P(50), nullptr, 0, 1024, N, 512, 1.f, 2);
  // 8. global max / mean pools
  pool_kernel<<<dim3(1024, Bn), 256, 0, stream>>>(buf_xx, buf_xmax, buf_xavg, 1024, N);
  // 9. class-label feature
  clf_kernel<<<1, 256, 0, stream>>>(cls, P(51), P(52), P(53), buf_clf);
  // 10. all_fea (B,3136,N)
  {
    long total = (long)Bn * 3136 * N;
    allfea_kernel<<<(int)((total + 255) / 256), 256, 0, stream>>>(
        buf_xx, buf_xmax, buf_xavg, buf_clf, buf_big, N);
  }

  float* sx = (float*)d_out;                          // (4,33,2048)
  float* sy = (float*)d_out + (size_t)Bn * 33 * N;    // (4,2,2048)

  // 11. head 1 -> sx
  gemm(P(54), 0, 3136, 1, buf_big, (long)3136 * N, N, buf_t1, (long)512 * N,
       P(55), P(56), P(57), nullptr, 0, 512, N, 3136, 1.f, 1);
  gemm(P(58), 0, 512, 1, buf_t1, (long)512 * N, N, buf_t2, (long)256 * N,
       P(59), P(60), P(61), nullptr, 0, 256, N, 512, 1.f, 1);
  gemm(P(62), 0, 256, 1, buf_t2, (long)256 * N, N, sx, (long)33 * N,
       P(63), nullptr, nullptr, nullptr, 0, 33, N, 256, 1.f, 0);
  // 12. head 2 -> sy  (reuse t1/t2)
  gemm(P(64), 0, 3136, 1, buf_big, (long)3136 * N, N, buf_t1, (long)512 * N,
       P(65), P(66), P(67), nullptr, 0, 512, N, 3136, 1.f, 1);
  gemm(P(68), 0, 512, 1, buf_t1, (long)512 * N, N, buf_t2, (long)256 * N,
       P(69), P(70), P(71), nullptr, 0, 256, N, 512, 1.f, 1);
  gemm(P(72), 0, 256, 1, buf_t2, (long)256 * N, N, sy, (long)2 * N,
       P(73), nullptr, nullptr, nullptr, 0, 2, N, 256, 1.f, 0);
}
